// MultiHeadAttention_44830868635939
// MI455X (gfx1250) — matrix operand order repaired and checked by reading, not compiled, not run on hardware
//
#include <hip/hip_runtime.h>

#define HIDDEN   2048
#define HEADS    16
#define HEAD_DIM 128
#define BATCH    2
#define SEQ      2048
#define MTOT     (BATCH * SEQ) /* 4096 */

typedef _Float16 half_t;
typedef __attribute__((ext_vector_type(8)))  _Float16 f16x8;
typedef __attribute__((ext_vector_type(16))) _Float16 f16x16;
typedef __attribute__((ext_vector_type(8)))  float    f32x8;
typedef __attribute__((ext_vector_type(4)))  unsigned int u32x4;
typedef __attribute__((ext_vector_type(4)))  int      i32x4;
typedef __attribute__((ext_vector_type(8)))  int      i32x8;

#if __has_builtin(__builtin_amdgcn_tensor_load_to_lds)
#define HAVE_TDM 1
#else
#define HAVE_TDM 0
#endif

// ---------------------------------------------------------------------------
// WMMA wrapper: D = A(16x32 f16) * B(32x16 f16) + C(16x16 f32)
// ---------------------------------------------------------------------------
__device__ __forceinline__ f32x8 wmma16(f16x16 a, f16x16 b, f32x8 c) {
  return __builtin_amdgcn_wmma_f32_16x16x32_f16(
      /*neg_a=*/false, a, /*neg_b=*/false, b,
      /*c_mod=*/(short)0, c, /*reuse_a=*/false, /*reuse_b=*/false);
}

__device__ __forceinline__ f16x16 join16(f16x8 lo, f16x8 hi) {
  f16x16 r;
#pragma unroll
  for (int i = 0; i < 8; ++i) { r[i] = lo[i]; r[i + 8] = hi[i]; }
  return r;
}

// A-matrix fragment (16x32 f16). Row m = lane%16.
// lanes 0-15: K = {k0+0..15}; lanes 16-31: K = {k0+16..31}.
__device__ __forceinline__ f16x16 load_a16(const half_t* base, int ld, int m0, int k0) {
  const int lane = threadIdx.x & 31;
  const half_t* p = base + (size_t)(m0 + (lane & 15)) * ld + k0 + ((lane < 16) ? 0 : 16);
  return join16(*(const f16x8*)p, *(const f16x8*)(p + 8));
}

// B-matrix fragment (32x16 f16) where B[k][n] = base[(row0+n)*ld + k0 + k].
__device__ __forceinline__ f16x16 load_bT(const half_t* base, int ld, int row0, int k0) {
  const int lane = threadIdx.x & 31;
  const half_t* p = base + (size_t)(row0 + (lane & 15)) * ld + k0 + ((lane < 16) ? 0 : 16);
  return join16(*(const f16x8*)p, *(const f16x8*)(p + 8));
}

// B-matrix fragment from a row-major [K][N] tile: B[k][n] = base[(k0+k)*ld + n0+n].
__device__ __forceinline__ f16x16 load_b_col(const half_t* base, int ld, int k0, int n0) {
  const int lane = threadIdx.x & 31;
  const int col = n0 + (lane & 15);
  const int kk = k0 + ((lane < 16) ? 0 : 16);
  f16x16 r;
#pragma unroll
  for (int h = 0; h < 16; ++h) r[h] = base[(size_t)(kk + h) * ld + col];
  return r;
}

#if HAVE_TDM
// ---------------------------------------------------------------------------
// TDM: async-load one 32-row x 128-col f16 tile (row stride 128 elems) to LDS.
// D# per CDNA5 ISA 8.3/8.4: group0 = {count=1, lds_addr, global_addr, type=2},
// group1 = {data_size=2B, tensor 128x32, tile 128x32, dim0_stride=128}.
// ---------------------------------------------------------------------------
__device__ __forceinline__ void tdm_load_tile(unsigned lds_off, const half_t* gptr) {
  unsigned long long ga = (unsigned long long)(size_t)(const void*)gptr;
  u32x4 g0 = { 1u,                                     // count=1, no gather
               lds_off,                                // lds_addr
               (unsigned)(ga & 0xffffffffu),           // global_addr[31:0]
               (unsigned)((ga >> 32) & 0x1ffffffu) | (2u << 30) }; // ga[56:32], type=2
  i32x8 g1 = { (int)0x00010000,                        // data_size=1 (2 bytes)
               (int)(128u << 16),                      // tensor_dim0 = 128
               (int)(32u << 16),                       // tensor_dim1 = 32
               (int)(128u << 16),                      // tile_dim0 = 128
               32,                                     // tile_dim1 = 32 (tile_dim2=0)
               128,                                    // tensor_dim0_stride = 128
               0, 0 };
  i32x4 z4 = { 0, 0, 0, 0 };
#if defined(__clang_major__) && (__clang_major__ >= 23)
  i32x8 z8 = { 0, 0, 0, 0, 0, 0, 0, 0 };
  __builtin_amdgcn_tensor_load_to_lds(g0, g1, z4, z4, z8, 0);
#else
  __builtin_amdgcn_tensor_load_to_lds(g0, g1, z4, z4, 0);
#endif
}
#endif

// ---------------------------------------------------------------------------
// fp32 -> f16 cast
// ---------------------------------------------------------------------------
__global__ void cast_f32_f16(const float* __restrict__ src, half_t* __restrict__ dst, int n) {
  int i = blockIdx.x * blockDim.x + threadIdx.x;
  if (i < n) dst[i] = (half_t)src[i];
}

// ---------------------------------------------------------------------------
// Projection GEMM: out[m, o] = sum_h X[m,h] * W[o,h]; f16 out in [B,H,S,D].
// One wave -> 32x64 tile: 2 A-frags x 4 B-frags = 8 WMMA per K step.
// ---------------------------------------------------------------------------
__global__ void gemm_qkv(const half_t* __restrict__ X, const half_t* __restrict__ W,
                         half_t* __restrict__ out_bhsd) {
  const int lane = threadIdx.x & 31;
  const int m0 = blockIdx.x * 32;
  const int o0 = blockIdx.y * 64;
  f32x8 acc[2][4] = {};
  for (int k = 0; k < HIDDEN; k += 32) {
    __builtin_prefetch(X + (size_t)(m0 + (lane & 15)) * HIDDEN + k + 256, 0, 1);
    __builtin_prefetch(W + (size_t)(o0 + (lane & 15)) * HIDDEN + k + 256, 0, 1);
    f16x16 a0 = load_a16(X, HIDDEN, m0, k);
    f16x16 a1 = load_a16(X, HIDDEN, m0 + 16, k);
#pragma unroll
    for (int ob = 0; ob < 4; ++ob) {
      f16x16 b = load_bT(W, HIDDEN, o0 + ob * 16, k);
      acc[0][ob] = wmma16(a0, b, acc[0][ob]);
      acc[1][ob] = wmma16(a1, b, acc[1][ob]);
    }
  }
  const int half8 = (lane < 16) ? 0 : 8;
  const int n = lane & 15;
#pragma unroll
  for (int mt = 0; mt < 2; ++mt)
#pragma unroll
    for (int ob = 0; ob < 4; ++ob)
#pragma unroll
      for (int r = 0; r < 8; ++r) {
        int m = m0 + mt * 16 + r + half8;
        int o = o0 + ob * 16 + n;
        int b = m >> 11, s = m & (SEQ - 1);
        int h = o >> 7,  d = o & (HEAD_DIM - 1);
        out_bhsd[(((size_t)b * HEADS + h) * SEQ + s) * HEAD_DIM + d] = (half_t)acc[mt][ob][r];
      }
}

// Output GEMM: d_out[m, o] = sum_h A[m,h] * Wo[o,h], fp32 result.
__global__ void gemm_out(const half_t* __restrict__ X, const half_t* __restrict__ W,
                         float* __restrict__ out) {
  const int lane = threadIdx.x & 31;
  const int m0 = blockIdx.x * 32;
  const int o0 = blockIdx.y * 64;
  f32x8 acc[2][4] = {};
  for (int k = 0; k < HIDDEN; k += 32) {
    __builtin_prefetch(X + (size_t)(m0 + (lane & 15)) * HIDDEN + k + 256, 0, 1);
    __builtin_prefetch(W + (size_t)(o0 + (lane & 15)) * HIDDEN + k + 256, 0, 1);
    f16x16 a0 = load_a16(X, HIDDEN, m0, k);
    f16x16 a1 = load_a16(X, HIDDEN, m0 + 16, k);
#pragma unroll
    for (int ob = 0; ob < 4; ++ob) {
      f16x16 b = load_bT(W, HIDDEN, o0 + ob * 16, k);
      acc[0][ob] = wmma16(a0, b, acc[0][ob]);
      acc[1][ob] = wmma16(a1, b, acc[1][ob]);
    }
  }
  const int half8 = (lane < 16) ? 0 : 8;
  const int n = lane & 15;
#pragma unroll
  for (int mt = 0; mt < 2; ++mt)
#pragma unroll
    for (int ob = 0; ob < 4; ++ob)
#pragma unroll
      for (int r = 0; r < 8; ++r)
        out[(size_t)(m0 + mt * 16 + r + half8) * HIDDEN + o0 + ob * 16 + n] = acc[mt][ob][r];
}

// ---------------------------------------------------------------------------
// RoPE in place on [B,H,S,D] f16. Each thread owns pair (dd, dd+64).
// ---------------------------------------------------------------------------
__global__ void rope_kernel(half_t* __restrict__ q) {
  int t = blockIdx.x * blockDim.x + threadIdx.x;
  int dd = t & 63;
  int s  = (t >> 6) & (SEQ - 1);
  int bh = t >> 17;
  if (bh >= BATCH * HEADS) return;
  half_t* row = q + ((size_t)bh * SEQ + s) * HEAD_DIM;
  float x1 = (float)row[dd];
  float x2 = (float)row[dd + 64];
  float inv = expf(-(float)dd * (9.2103403719761836f / 64.0f)); // 10000^(-dd/64)
  float ang = (float)s * inv;
  float c = cosf(ang), si = sinf(ang);
  row[dd]      = (half_t)(x1 * c - x2 * si);
  row[dd + 64] = (half_t)(x2 * c + x1 * si);
}

// ---------------------------------------------------------------------------
// Flash attention, causal. 4 waves per block; each wave owns a 16-query tile
// of a 64-row block. Wave 0 drives the TDM double-buffered K/V staging
// (s_wait_tensorcnt), workgroup barriers publish tiles to all 4 waves.
// ---------------------------------------------------------------------------
__global__ void flash_attn(const half_t* __restrict__ qh, const half_t* __restrict__ kh,
                           const half_t* __restrict__ vh, half_t* __restrict__ oh) {
  __shared__ half_t lds_k[2][32 * HEAD_DIM];   // 2 x 8 KB
  __shared__ half_t lds_v[2][32 * HEAD_DIM];   // 2 x 8 KB
  __shared__ half_t lds_p[4][16 * 32];         // per-wave P tile, 4 x 1 KB

  const int lane  = threadIdx.x & 31;
  const int wave  = threadIdx.x >> 5;
  const int bh    = blockIdx.y;
  const int q0    = blockIdx.x * 64 + wave * 16;  // this wave's query rows
  const int bqmax = blockIdx.x * 64 + 63;         // block's last query row
  const int half8 = (lane < 16) ? 0 : 8;
  const int n     = lane & 15;

  const half_t* qb = qh + (size_t)bh * SEQ * HEAD_DIM;
  const half_t* kb = kh + (size_t)bh * SEQ * HEAD_DIM;
  const half_t* vb = vh + (size_t)bh * SEQ * HEAD_DIM;

  f16x16 qf[4];
#pragma unroll
  for (int kc = 0; kc < 4; ++kc) qf[kc] = load_a16(qb, HEAD_DIM, q0, kc * 32);

  f32x8 o[8] = {};
  float m8[8], l8[8];
#pragma unroll
  for (int r = 0; r < 8; ++r) { m8[r] = -1e30f; l8[r] = 0.0f; }

  const float scale = 0.08838834764831845f;    // 1/sqrt(128)

#if HAVE_TDM
  if (wave == 0) {
    tdm_load_tile((unsigned)(size_t)(void*)&lds_k[0][0], kb);
    tdm_load_tile((unsigned)(size_t)(void*)&lds_v[0][0], vb);
  }
#endif

  for (int j0 = 0; j0 <= bqmax; j0 += 32) {
    const int cur = (j0 >> 5) & 1;
#if HAVE_TDM
    if (wave == 0) {
      if (j0 + 32 <= bqmax) {
        tdm_load_tile((unsigned)(size_t)(void*)&lds_k[cur ^ 1][0],
                      kb + (size_t)(j0 + 32) * HEAD_DIM);
        tdm_load_tile((unsigned)(size_t)(void*)&lds_v[cur ^ 1][0],
                      vb + (size_t)(j0 + 32) * HEAD_DIM);
        __builtin_amdgcn_s_wait_tensorcnt(2);  // current pair done (in-order)
      } else {
        __builtin_amdgcn_s_wait_tensorcnt(0);
      }
    }
    __syncthreads();                            // publish current tile to all waves
#else
    // Cooperative staging: 512 x f16x8 chunks per tile across 128 threads.
#pragma unroll
    for (int i = 0; i < 4; ++i) {
      int ch  = threadIdx.x + i * 128;          // 0..511
      int row = ch >> 4, off = (ch & 15) * 8;
      *(f16x8*)(&lds_k[cur][row * HEAD_DIM + off]) =
          *(const f16x8*)(kb + (size_t)(j0 + row) * HEAD_DIM + off);
      *(f16x8*)(&lds_v[cur][row * HEAD_DIM + off]) =
          *(const f16x8*)(vb + (size_t)(j0 + row) * HEAD_DIM + off);
    }
    __syncthreads();
#endif

    // ---- scores S[16 x 32] from LDS K-tile ----
    f32x8 c0 = {}, c1 = {};
#pragma unroll
    for (int kc = 0; kc < 4; ++kc) {
      f16x16 b0 = load_bT(&lds_k[cur][0], HEAD_DIM, 0,  kc * 32);
      f16x16 b1 = load_bT(&lds_k[cur][0], HEAD_DIM, 16, kc * 32);
      c0 = wmma16(qf[kc], b0, c0);
      c1 = wmma16(qf[kc], b1, c1);
    }

    // ---- online softmax (fully-masked tiles degrade to alpha=1, p=0) ----
    float alpha[8];
#pragma unroll
    for (int r = 0; r < 8; ++r) {
      int row = q0 + r + half8;
      float s0 = c0[r] * scale; if (j0 + n      > row) s0 = -1e30f;
      float s1 = c1[r] * scale; if (j0 + 16 + n > row) s1 = -1e30f;
      float mx = fmaxf(s0, s1);
#pragma unroll
      for (int off = 1; off < 16; off <<= 1) mx = fmaxf(mx, __shfl_xor(mx, off, 32));
      float mnew = fmaxf(m8[r], mx);
      float p0 = __expf(s0 - mnew);
      float p1 = __expf(s1 - mnew);
      float rs = p0 + p1;
#pragma unroll
      for (int off = 1; off < 16; off <<= 1) rs += __shfl_xor(rs, off, 32);
      alpha[r] = __expf(m8[r] - mnew);
      l8[r] = l8[r] * alpha[r] + rs;
      m8[r] = mnew;
      lds_p[wave][(r + half8) * 32 + n]      = (half_t)p0;
      lds_p[wave][(r + half8) * 32 + 16 + n] = (half_t)p1;
    }

#pragma unroll
    for (int f = 0; f < 8; ++f)
#pragma unroll
      for (int r = 0; r < 8; ++r) o[f][r] *= alpha[r];

    // ---- O += P * V (P in-wave through LDS; V column-wise from LDS tile) ----
    f16x16 pf = load_a16(&lds_p[wave][0], 32, 0, 0);
#pragma unroll
    for (int f = 0; f < 8; ++f) {
      f16x16 bv = load_b_col(&lds_v[cur][0], HEAD_DIM, 0, f * 16);
      o[f] = wmma16(pf, bv, o[f]);
    }
    __syncthreads();   // all waves done with this buffer before it is re-filled
  }

  // ---- normalize and write [B,S,H*D] f16 ----
  const int b = bh / HEADS, hd = bh % HEADS;
#pragma unroll
  for (int f = 0; f < 8; ++f)
#pragma unroll
    for (int r = 0; r < 8; ++r) {
      int srow = q0 + r + half8;
      oh[((size_t)b * SEQ + srow) * HIDDEN + hd * HEAD_DIM + f * 16 + n] =
          (half_t)(o[f][r] / l8[r]);
    }
}

// ---------------------------------------------------------------------------
// Launch
// ---------------------------------------------------------------------------
extern "C" void kernel_launch(void* const* d_in, const int* in_sizes, int n_in,
                              void* d_out, int out_size, void* d_ws, size_t ws_size,
                              hipStream_t stream) {
  (void)in_sizes; (void)n_in; (void)out_size; (void)ws_size;
  const float* x  = (const float*)d_in[0];
  const float* Wq = (const float*)d_in[1];
  const float* Wk = (const float*)d_in[2];
  const float* Wv = (const float*)d_in[3];
  const float* Wo = (const float*)d_in[4];

  const size_t XN = (size_t)MTOT * HIDDEN;
  const size_t WN = (size_t)HIDDEN * HIDDEN;

  char* w = (char*)d_ws;
  half_t* xh  = (half_t*)w; w += XN * 2;
  half_t* Wqh = (half_t*)w; w += WN * 2;
  half_t* Wkh = (half_t*)w; w += WN * 2;
  half_t* Wvh = (half_t*)w; w += WN * 2;
  half_t* Woh = (half_t*)w; w += WN * 2;
  half_t* qhb = (half_t*)w; w += XN * 2;       // [B,H,S,D]
  half_t* khb = (half_t*)w; w += XN * 2;
  half_t* vhb = (half_t*)w; w += XN * 2;
  half_t* ah  = (half_t*)w; w += XN * 2;       // attention out, [B,S,H*D]

  cast_f32_f16<<<(unsigned)((XN + 255) / 256), 256, 0, stream>>>(x,  xh,  (int)XN);
  cast_f32_f16<<<(unsigned)((WN + 255) / 256), 256, 0, stream>>>(Wq, Wqh, (int)WN);
  cast_f32_f16<<<(unsigned)((WN + 255) / 256), 256, 0, stream>>>(Wk, Wkh, (int)WN);
  cast_f32_f16<<<(unsigned)((WN + 255) / 256), 256, 0, stream>>>(Wv, Wvh, (int)WN);
  cast_f32_f16<<<(unsigned)((WN + 255) / 256), 256, 0, stream>>>(Wo, Woh, (int)WN);

  dim3 gg(MTOT / 32, HIDDEN / 64);             // (128, 32), 32 threads = 1 wave
  gemm_qkv<<<gg, 32, 0, stream>>>(xh, Wqh, qhb);
  gemm_qkv<<<gg, 32, 0, stream>>>(xh, Wkh, khb);
  gemm_qkv<<<gg, 32, 0, stream>>>(xh, Wvh, vhb);

  const int rope_total = BATCH * HEADS * SEQ * 64;
  rope_kernel<<<rope_total / 256, 256, 0, stream>>>(qhb);
  rope_kernel<<<rope_total / 256, 256, 0, stream>>>(khb);

  flash_attn<<<dim3(SEQ / 64, BATCH * HEADS), 128, 0, stream>>>(qhb, khb, vhb, ah);

  gemm_out<<<gg, 32, 0, stream>>>(ah, Woh, (float*)d_out);
}
